// TopicSNE_8332236554543
// MI455X (gfx1250) — matrix operand order repaired and checked by reading, not compiled, not run on hardware
//
#include <hip/hip_runtime.h>
#include <hip/hip_bf16.h>

typedef __attribute__((ext_vector_type(16))) _Float16 v16h;
typedef __attribute__((ext_vector_type(8)))  _Float16 v8h;
typedef __attribute__((ext_vector_type(8)))  float    v8f;

#define N_OBS   8192
#define N_TOP   128
#define N_DIM   64
#define N_BATCH 8192
#define NSPLIT  32
#define NCHUNK  (N_OBS / NSPLIT)   // 256 -> 16 N-steps per wave

// ---------------------------------------------------------------------------
// helpers
// ---------------------------------------------------------------------------
__device__ __forceinline__ float wave_sum(float v) {
#pragma unroll
    for (int off = 16; off > 0; off >>= 1) v += __shfl_xor(v, off, 32);
    return v;
}
__device__ __forceinline__ float wave_max(float v) {
#pragma unroll
    for (int off = 16; off > 0; off >>= 1) v = fmaxf(v, __shfl_xor(v, off, 32));
    return v;
}

// concat two 16B chunks into a v16h (two global_load_b128)
__device__ __forceinline__ v16h load16h(const _Float16* p0, const _Float16* p1) {
    v8h lo = *(const v8h*)p0;
    v8h hi = *(const v8h*)p1;
    v16h r;
#pragma unroll
    for (int k = 0; k < 8; ++k) { r[k] = lo[k]; r[k + 8] = hi[k]; }
    return r;
}

// One row of  x = softmax(logits + gumbel(noise)) @ W^T + b.
// Wave-cooperative: 128 topics = 4 per lane; s staged via per-wave LDS slice
// (same-wave LDS ops are in-order on CDNA5). Wt is W transposed in LDS
// (Wt[t*64+d]) so GEMV reads are bank-conflict free. Each lane returns
// dims d=lane and d=lane+32.
__device__ __forceinline__ void topic_row(const float* __restrict__ logits_row,
                                          const float* __restrict__ noise_row,
                                          const float* Wt, float* sbuf,
                                          const float* __restrict__ topic_b,
                                          int lane, float& x0, float& x1) {
    float z[4];
#pragma unroll
    for (int c = 0; c < 4; ++c) {
        int t = lane + c * 32;
        float u = noise_row[t];
        float g = -__logf(-__logf(u + 1e-9f) + 1e-9f);   // gumbel
        z[c] = logits_row[t] + g;
    }
    float m = fmaxf(fmaxf(z[0], z[1]), fmaxf(z[2], z[3]));
    m = wave_max(m);
    float e[4], s = 0.f;
#pragma unroll
    for (int c = 0; c < 4; ++c) { e[c] = __expf(z[c] - m); s += e[c]; }
    s = wave_sum(s);
    float inv = __builtin_amdgcn_rcpf(s);
#pragma unroll
    for (int c = 0; c < 4; ++c) sbuf[lane + c * 32] = e[c] * inv;

    x0 = topic_b[lane];
    x1 = topic_b[lane + 32];
#pragma unroll 8
    for (int t = 0; t < N_TOP; ++t) {
        float sv = sbuf[t];                       // broadcast
        x0 = fmaf(sv, Wt[t * N_DIM + lane], x0);
        x1 = fmaf(sv, Wt[t * N_DIM + lane + 32], x1);
    }
}

// ---------------------------------------------------------------------------
// Kernel 1: build x (f16) + row norms + zero the part accumulator
// ---------------------------------------------------------------------------
__global__ __launch_bounds__(256) void build_x_kernel(
    const float* __restrict__ logits, const float* __restrict__ noise_full,
    const float* __restrict__ topic_w, const float* __restrict__ topic_b,
    _Float16* __restrict__ xh, float* __restrict__ sq, float* __restrict__ part) {
    __shared__ float Wt[N_TOP * N_DIM];    // 32 KB, transposed W
    __shared__ float sbuf[8 * N_TOP];      // 4 KB, softmax stage per wave
    const int tid = threadIdx.x;
    for (int e = tid; e < N_TOP * N_DIM; e += 256) {
        int t = e >> 6, d = e & 63;
        Wt[e] = topic_w[d * N_TOP + t];
    }
    __syncthreads();

    const int wave = tid >> 5, lane = tid & 31;
    const int row = blockIdx.x * 8 + wave;

    float x0, x1;
    topic_row(logits + (size_t)row * N_TOP, noise_full + (size_t)row * N_TOP,
              Wt, sbuf + wave * N_TOP, topic_b, lane, x0, x1);

    xh[(size_t)row * N_DIM + lane]      = (_Float16)x0;
    xh[(size_t)row * N_DIM + lane + 32] = (_Float16)x1;

    float s2 = wave_sum(x0 * x0 + x1 * x1);
    if (lane == 0) sq[row] = s2;
    if (blockIdx.x == 0 && tid == 0) *part = 0.f;
}

// ---------------------------------------------------------------------------
// Kernel 2: part = sum over 8192x8192 of 1/(1 + sq_i + sq_j - 2*x_i.x_j)
// 1 wave per block; 64 rows x 256 cols per wave; v_wmma_f32_16x16x32_f16.
// ---------------------------------------------------------------------------
__global__ __launch_bounds__(32) void pairwise_kernel(
    const _Float16* __restrict__ xh, const float* __restrict__ sq,
    float* __restrict__ part) {
    const int lane = threadIdx.x & 31;
    const int hi = lane >> 4;          // half-wave select
    const int lo = lane & 15;
    const int m0 = blockIdx.x * 64;
    const int n0 = blockIdx.y * NCHUNK;

    // A operand: lane (lo,hi) holds row m0+mt*16+lo, K = kstep*32 + hi*8 + {0..7,16..23}
    v16h A[4][2];
    float sqA[4][8];
#pragma unroll
    for (int mt = 0; mt < 4; ++mt) {
        const _Float16* pr = xh + (size_t)(m0 + mt * 16 + lo) * N_DIM + hi * 8;
        A[mt][0] = load16h(pr,      pr + 16);
        A[mt][1] = load16h(pr + 32, pr + 48);
#pragma unroll
        for (int e = 0; e < 8; ++e) sqA[mt][e] = sq[m0 + mt * 16 + hi * 8 + e];
    }

    float psum = 0.f;
#pragma unroll 2
    for (int n = n0; n < n0 + NCHUNK; n += 16) {
        // B operand (= X^T tile): lane (lo,hi) holds row n+lo, K = kstep*32 + hi*8 + {0..7,16..23}
        const _Float16* pb = xh + (size_t)(n + lo) * N_DIM + hi * 8;
        v16h B0 = load16h(pb,      pb + 16);
        v16h B1 = load16h(pb + 32, pb + 48);
        float sqj1 = sq[n + lo] + 1.0f;    // hoist the "+1" of 1/(1+d)

#pragma unroll
        for (int mt = 0; mt < 4; ++mt) {
            v8f c = {};
            c = __builtin_amdgcn_wmma_f32_16x16x32_f16(false, A[mt][0], false, B0,
                                                       (short)0, c, false, false);
            c = __builtin_amdgcn_wmma_f32_16x16x32_f16(false, A[mt][1], false, B1,
                                                       (short)0, c, false, false);
#pragma unroll
            for (int e = 0; e < 8; ++e) {
                // 1 + dkl2 = (sqA + (sqj+1)) - 2*G   -> pk_add + pk_fma + rcp
                float t = fmaf(-2.0f, c[e], sqA[mt][e] + sqj1);
                psum += __builtin_amdgcn_rcpf(t);
            }
        }
    }
    psum = wave_sum(psum);
    if (lane == 0) atomicAdd(part, psum);
}

// ---------------------------------------------------------------------------
// Kernel 3: per-pair loss = pij*(log pij + log(64 + ||xi-xj||^2) + log(part))
// ---------------------------------------------------------------------------
__global__ __launch_bounds__(256) void batch_kernel(
    const float* __restrict__ pij, const float* __restrict__ noise_i,
    const float* __restrict__ noise_j, const float* __restrict__ logits,
    const float* __restrict__ topic_w, const float* __restrict__ topic_b,
    const int* __restrict__ iv, const int* __restrict__ jv,
    const float* __restrict__ part, float* __restrict__ out) {
    __shared__ float Wt[N_TOP * N_DIM];
    __shared__ float sbuf[8 * N_TOP];
    const int tid = threadIdx.x;
    for (int e = tid; e < N_TOP * N_DIM; e += 256) {
        int t = e >> 6, d = e & 63;
        Wt[e] = topic_w[d * N_TOP + t];
    }
    __syncthreads();

    const int wave = tid >> 5, lane = tid & 31;
    const int b = blockIdx.x * 8 + wave;
    const int ii = iv[b], jj = jv[b];

    float xi0, xi1, xj0, xj1;
    topic_row(logits + (size_t)ii * N_TOP, noise_i + (size_t)b * N_TOP,
              Wt, sbuf + wave * N_TOP, topic_b, lane, xi0, xi1);
    topic_row(logits + (size_t)jj * N_TOP, noise_j + (size_t)b * N_TOP,
              Wt, sbuf + wave * N_TOP, topic_b, lane, xj0, xj1);

    float d0 = xi0 - xj0, d1 = xi1 - xj1;
    float d2 = wave_sum(d0 * d0 + d1 * d1);
    if (lane == 0) {
        float p = pij[b];
        float partv = *part - (float)N_OBS;
        // loss = p*(log p - log(qij)), qij = 1/(64+d2) / part
        out[b] = p * (__logf(p) + __logf((float)N_DIM + d2) + __logf(partv));
    }
}

// ---------------------------------------------------------------------------
extern "C" void kernel_launch(void* const* d_in, const int* in_sizes, int n_in,
                              void* d_out, int out_size, void* d_ws, size_t ws_size,
                              hipStream_t stream) {
    const float* pij        = (const float*)d_in[0];
    const float* noise_full = (const float*)d_in[1];
    const float* noise_i    = (const float*)d_in[2];
    const float* noise_j    = (const float*)d_in[3];
    const float* logits     = (const float*)d_in[4];
    const float* topic_w    = (const float*)d_in[5];
    const float* topic_b    = (const float*)d_in[6];
    const int*   iv         = (const int*)d_in[7];
    const int*   jv         = (const int*)d_in[8];
    float* out = (float*)d_out;

    char* ws = (char*)d_ws;
    _Float16* xh = (_Float16*)ws;                                   // 8192*64 f16 = 1 MB
    float* sq    = (float*)(ws + (size_t)N_OBS * N_DIM * 2);        // 8192 f32
    float* part  = sq + N_OBS;                                      // 1 f32

    build_x_kernel<<<N_OBS / 8, 256, 0, stream>>>(logits, noise_full, topic_w,
                                                  topic_b, xh, sq, part);
    dim3 g2(N_OBS / 64, NSPLIT);
    pairwise_kernel<<<g2, 32, 0, stream>>>(xh, sq, part);
    batch_kernel<<<N_BATCH / 8, 256, 0, stream>>>(pij, noise_i, noise_j, logits,
                                                  topic_w, topic_b, iv, jv, part, out);
}
